// GraphSAGE_30829275250829
// MI455X (gfx1250) — compile-verified, matrix operand order checked by reading, not run on hardware
//
#include <hip/hip_runtime.h>
#include <math.h>

#define NNODES 50000
#define NEDGES 800000
#define DIN    128
#define DH     128
#define DOUTF  64
#define KTOT   256   // concat K = DIN (mean) + DIN (self)
#define NKT    8     // KTOT / 32

typedef __attribute__((ext_vector_type(16))) __bf16 v16bf;
typedef __attribute__((ext_vector_type(8)))  float  v8f;

// f32 -> bf16 round-to-nearest-even (bit pattern)
static __device__ __forceinline__ unsigned short f2bf_bits(float f) {
  union { float f; unsigned u; } v; v.f = f;
  unsigned r = v.u + 0x7fffu + ((v.u >> 16) & 1u);
  return (unsigned short)(r >> 16);
}
static __device__ __forceinline__ unsigned packbf(float lo, float hi) {
  return (unsigned)f2bf_bits(lo) | ((unsigned)f2bf_bits(hi) << 16);
}
// K offset within a 32-wide K tile for fragment register i, given lane half.
static __device__ __forceinline__ int frag_koff(int kbase, int i) {
  return (i < 4) ? (kbase + 2 * i) : (kbase + 16 + 2 * (i - 4));
}

__global__ __launch_bounds__(256)
void zero_f32(float* __restrict__ p, long long n) {
  long long i = (long long)blockIdx.x * blockDim.x + threadIdx.x;
  long long stride = (long long)gridDim.x * blockDim.x;
  for (; i < n; i += stride) p[i] = 0.0f;
}

__global__ __launch_bounds__(256)
void deg_count(const int* __restrict__ dst, float* __restrict__ deg, int e) {
  int i = blockIdx.x * blockDim.x + threadIdx.x;
  if (i < e) atomicAdd(&deg[dst[i]], 1.0f);
}

__global__ __launch_bounds__(256)
void deg_finalize(float* __restrict__ deg, int n) {
  int i = blockIdx.x * blockDim.x + threadIdx.x;
  if (i < n) deg[i] = 1.0f / fmaxf(deg[i], 1.0f);
}

// One edge handled by 32 lanes; each lane moves 4 consecutive feats.
__global__ __launch_bounds__(256)
void scatter_sum(const float* __restrict__ h, const int* __restrict__ src,
                 const int* __restrict__ dst, float* __restrict__ agg, int e) {
  int t = blockIdx.x * blockDim.x + threadIdx.x;
  int edge = t >> 5;
  if (edge >= e) return;
  int c4 = (t & 31) * 4;
  const float4 v = *(const float4*)(h + (long long)src[edge] * DIN + c4);
  float* a = agg + (long long)dst[edge] * DIN + c4;
  atomicAdd(a + 0, v.x);
  atomicAdd(a + 1, v.y);
  atomicAdd(a + 2, v.z);
  atomicAdd(a + 3, v.w);
}

// Pre-swizzle [Wl;Wr] (f32) into B-fragment layout:
// Bpack[((ct*NKT + kt)*32 + lane)*8 + i] = bf16x2( W[k][ct*16+(lane&15)] , W[k+1][...] )
template <int DOUT_T>
__global__ __launch_bounds__(256)
void pack_weights(const float* __restrict__ Wl, const float* __restrict__ Wr,
                  unsigned* __restrict__ Bpack) {
  const int total = (DOUT_T / 16) * NKT * 32 * 8;
  int idx = blockIdx.x * blockDim.x + threadIdx.x;
  if (idx >= total) return;
  int i    = idx & 7;
  int lane = (idx >> 3) & 31;
  int kt   = (idx >> 8) & (NKT - 1);
  int ct   = idx >> 11;
  int kbase = (lane < 16) ? 0 : 8;
  int k = kt * 32 + frag_koff(kbase, i);     // even; k,k+1 in same half
  int col = ct * 16 + (lane & 15);
  const float* W = (k < DIN) ? Wl : Wr;
  int kk = k & (DIN - 1);
  Bpack[idx] = packbf(W[kk * DOUT_T + col], W[(kk + 1) * DOUT_T + col]);
}

// out = ELU( [agg*invdeg | h] @ [Wl ; Wr] + b )
// One block per 16 rows; one wave32 per 16-col tile; A panel staged in LDS
// pre-swizzled so each K-step is 2x ds_load_b128 + 2x global_load_b128 + 1 WMMA.
template <int DOUT_T>
__global__ __launch_bounds__(32 * (DOUT_T / 16))
void sage_wmma(const float* __restrict__ h, const float* __restrict__ agg,
               const float* __restrict__ invdeg,
               const unsigned* __restrict__ Bpack,
               const float* __restrict__ bias, float* __restrict__ out, int n) {
  __shared__ __align__(16) unsigned ldsA[NKT * 32 * 8];   // 8 KB, [kt][lane][i]

  const int tid = threadIdx.x;
  const int rowBase = blockIdx.x * 16;

  // Stage A fragments: entry (kt,l,i) = bf16x2 of A[m][k], A[m][k+1],
  // where A = [mean | self] (K = 256), m = l&15, half split by l<16.
  for (int idx = tid; idx < NKT * 32 * 8; idx += blockDim.x) {
    int i  = idx & 7;
    int l  = (idx >> 3) & 31;
    int kt = idx >> 8;
    int m  = l & 15;
    int kbase = (l < 16) ? 0 : 8;
    int k = kt * 32 + frag_koff(kbase, i);
    int row = rowBase + m;
    float v0 = 0.0f, v1 = 0.0f;
    if (row < n) {
      if (k < DIN) {
        float2 p = *(const float2*)(agg + (long long)row * DIN + k);
        float inv = invdeg[row];
        v0 = p.x * inv; v1 = p.y * inv;
      } else {
        float2 p = *(const float2*)(h + (long long)row * DIN + (k - DIN));
        v0 = p.x; v1 = p.y;
      }
    }
    ldsA[idx] = packbf(v0, v1);
  }
  __syncthreads();

  const int wave = tid >> 5;
  const int lane = tid & 31;
  const int col  = wave * 16 + (lane & 15);

  union { v16bf v; uint4 q[2]; } afrag, bfrag;
  union { v8f v; float f[8]; } acc;
#pragma unroll
  for (int i = 0; i < 8; ++i) acc.f[i] = 0.0f;

  const uint4* A4 = (const uint4*)ldsA;
  const uint4* B4 = (const uint4*)Bpack;

#pragma unroll
  for (int kt = 0; kt < NKT; ++kt) {
    int ai = (kt * 32 + lane) * 2;
    afrag.q[0] = A4[ai];
    afrag.q[1] = A4[ai + 1];
    int bi = ((wave * NKT + kt) * 32 + lane) * 2;
    bfrag.q[0] = B4[bi];
    bfrag.q[1] = B4[bi + 1];
    acc.v = __builtin_amdgcn_wmma_f32_16x16x32_bf16(
        /*neg_a=*/false, afrag.v, /*neg_b=*/false, bfrag.v,
        /*c_mod=*/(short)0, acc.v, /*reuse_a=*/false, /*reuse_b=*/false);
  }

  // Epilogue: bias + ELU. C layout: VGPR r -> row (r + 8*(lane>=16)), col = lane&15.
  const float bv = bias[col];
  const int rowOff = (lane < 16) ? 0 : 8;
#pragma unroll
  for (int r = 0; r < 8; ++r) {
    int row = rowBase + rowOff + r;
    if (row < n) {
      float v = acc.f[r] + bv;
      v = (v > 0.0f) ? v : (expf(v) - 1.0f);   // ELU(alpha=1)
      out[(long long)row * DOUT_T + col] = v;
    }
  }
}

// In-place log_softmax over DOUTF=64 columns; one wave32 per row.
__global__ __launch_bounds__(256)
void log_softmax64(float* __restrict__ p, int n) {
  int wave = threadIdx.x >> 5;
  int lane = threadIdx.x & 31;
  int row = blockIdx.x * 8 + wave;
  if (row >= n) return;
  float* pr = p + (long long)row * DOUTF;
  float v0 = pr[lane];
  float v1 = pr[lane + 32];
  float m = fmaxf(v0, v1);
#pragma unroll
  for (int off = 16; off > 0; off >>= 1)
    m = fmaxf(m, __shfl_xor(m, off, 32));
  float s = expf(v0 - m) + expf(v1 - m);
#pragma unroll
  for (int off = 16; off > 0; off >>= 1)
    s += __shfl_xor(s, off, 32);
  float ls = logf(s) + m;
  pr[lane]      = v0 - ls;
  pr[lane + 32] = v1 - ls;
}

extern "C" void kernel_launch(void* const* d_in, const int* in_sizes, int n_in,
                              void* d_out, int out_size, void* d_ws, size_t ws_size,
                              hipStream_t stream) {
  (void)in_sizes; (void)n_in; (void)out_size; (void)ws_size;

  const float* x   = (const float*)d_in[0];
  const float* W1l = (const float*)d_in[1];
  const float* W1r = (const float*)d_in[2];
  const float* b1  = (const float*)d_in[3];
  const float* W2l = (const float*)d_in[4];
  const float* W2r = (const float*)d_in[5];
  const float* b2  = (const float*)d_in[6];
  const float* W3l = (const float*)d_in[7];
  const float* W3r = (const float*)d_in[8];
  const float* b3  = (const float*)d_in[9];
  const int*   src = (const int*)d_in[10];
  const int*   dst = (const int*)d_in[11];
  float* out = (float*)d_out;

  const int n = NNODES, e = NEDGES;

  // Workspace (floats): deg[N] | agg[N*128] | h1[N*128] | h2[N*128] | Bp1 | Bp2 | Bp3
  float* deg = (float*)d_ws;
  float* agg = deg + n;
  float* h1  = agg + (long long)n * DIN;
  float* h2  = h1  + (long long)n * DIN;
  unsigned* Bp1 = (unsigned*)(h2 + (long long)n * DIN);
  unsigned* Bp2 = Bp1 + (DH   / 16) * NKT * 32 * 8;   // 16384 u32
  unsigned* Bp3 = Bp2 + (DH   / 16) * NKT * 32 * 8;   // 16384 u32

  const int mtiles = (n + 15) / 16;
  const int scatterBlocks = (e * 32 + 255) / 256;
  const long long aggElems = (long long)n * DIN;

  // One-shot weight packing (constant across row-blocks)
  pack_weights<DH>   <<<(DH / 16)   * NKT * 32 * 8 / 256, 256, 0, stream>>>(W1l, W1r, Bp1);
  pack_weights<DH>   <<<(DH / 16)   * NKT * 32 * 8 / 256, 256, 0, stream>>>(W2l, W2r, Bp2);
  pack_weights<DOUTF><<<(DOUTF / 16) * NKT * 32 * 8 / 256, 256, 0, stream>>>(W3l, W3r, Bp3);

  // Degrees -> inverse degrees (shared by all layers)
  zero_f32<<<512, 256, 0, stream>>>(deg, n);
  deg_count<<<(e + 255) / 256, 256, 0, stream>>>(dst, deg, e);
  deg_finalize<<<(n + 255) / 256, 256, 0, stream>>>(deg, n);

  // Layer 1: x -> h1
  zero_f32<<<4096, 256, 0, stream>>>(agg, aggElems);
  scatter_sum<<<scatterBlocks, 256, 0, stream>>>(x, src, dst, agg, e);
  sage_wmma<DH><<<mtiles, 32 * (DH / 16), 0, stream>>>(x, agg, deg, Bp1, b1, h1, n);

  // Layer 2: h1 -> h2
  zero_f32<<<4096, 256, 0, stream>>>(agg, aggElems);
  scatter_sum<<<scatterBlocks, 256, 0, stream>>>(h1, src, dst, agg, e);
  sage_wmma<DH><<<mtiles, 32 * (DH / 16), 0, stream>>>(h1, agg, deg, Bp2, b2, h2, n);

  // Layer 3: h2 -> out (ELU applied), then log_softmax in place
  zero_f32<<<4096, 256, 0, stream>>>(agg, aggElems);
  scatter_sum<<<scatterBlocks, 256, 0, stream>>>(h2, src, dst, agg, e);
  sage_wmma<DOUTF><<<mtiles, 32 * (DOUTF / 16), 0, stream>>>(h2, agg, deg, Bp3, b3, out, n);

  log_softmax64<<<(n + 7) / 8, 256, 0, stream>>>(out, n);
}
